// SmolLM3Attention_2439541424354
// MI455X (gfx1250) — matrix operand order repaired and checked by reading, not compiled, not run on hardware
//
#include <hip/hip_runtime.h>
#include <hip/hip_bf16.h>

// ---------------------------------------------------------------- constants
#define B_ 2
#define S_ 2048
#define HID_ 2048
#define NH_ 16
#define NKV_ 4
#define HD_ 128
#define GROUPS_ (NH_ / NKV_)
#define SCALING_ 0.08838834764831845f   // 128^-0.5

typedef __bf16 bf16_t;
typedef __attribute__((ext_vector_type(16))) __bf16 v16bf;
typedef __attribute__((ext_vector_type(8)))  __bf16 bf16x8;
typedef __attribute__((ext_vector_type(8)))  float  v8f;
typedef int v4i_16B __attribute__((vector_size(16)));   // int4, matches builtin proto

#if __has_builtin(__builtin_amdgcn_global_load_async_to_lds_b128)
#define HAS_ASYNC_LDS 1
#else
#define HAS_ASYNC_LDS 0
#endif

// ---------------------------------------------------------------- helpers
__device__ __forceinline__ bf16_t f2bf(float f) {
  union { float f; unsigned u; } x; x.f = f;
  unsigned r = x.u + 0x7FFFu + ((x.u >> 16) & 1u);   // round-to-nearest-even
  unsigned short h = (unsigned short)(r >> 16);
  return __builtin_bit_cast(bf16_t, h);
}

__device__ __forceinline__ v8f wmma_bf16(v16bf a, v16bf b, v8f c) {
  // emits v_wmma_f32_16x16x32_bf16
  return __builtin_amdgcn_wmma_f32_16x16x32_bf16(
      /*neg_a=*/false, a, /*neg_b=*/false, b,
      /*c_mod=*/(short)0, c, /*reuse_a=*/false, /*reuse_b=*/false);
}

// 16-byte global -> LDS copy. Async (ASYNCcnt / global_load_async_to_lds_b128)
// when the toolchain exposes the gfx1250 builtin, else synchronous via regs.
__device__ __forceinline__ void cp16_g2l(const bf16_t* g, bf16_t* l) {
#if HAS_ASYNC_LDS
  __builtin_amdgcn_global_load_async_to_lds_b128(
      (__attribute__((address_space(1))) v4i_16B*)(uintptr_t)g,
      (__attribute__((address_space(3))) v4i_16B*)(unsigned)(uintptr_t)l,
      0, 0);
#else
  *(bf16x8*)l = *(const bf16x8*)g;
#endif
}

__device__ __forceinline__ void wait_async_lds() {
#if HAS_ASYNC_LDS
#if __has_builtin(__builtin_amdgcn_s_wait_asynccnt)
  __builtin_amdgcn_s_wait_asynccnt(0);
#else
  asm volatile("s_wait_asynccnt 0" ::: "memory");
#endif
#endif
}

// A fragment: 16x32 bf16, A row-major (lda), rows base..base+15, cols k0..k0+31.
// Lane L: row = L&15, khalf = (L>>4)*8. Element e (0..15):
//   e<8 : k = khalf + e        e>=8: k = 16 + khalf + (e-8)
__device__ __forceinline__ v16bf load_a16(const bf16_t* __restrict__ A, int lda,
                                          int lane, int k0) {
  int r  = lane & 15;
  int kh = (lane >> 4) * 8;
  const bf16_t* p = A + (size_t)r * lda + k0 + kh;
  bf16x8 lo = *(const bf16x8*)(p);
  bf16x8 hi = *(const bf16x8*)(p + 16);
  v16bf out;
#pragma unroll
  for (int i = 0; i < 8; ++i) { out[i] = lo[i]; out[i + 8] = hi[i]; }
  return out;
}

// Same A fragment but source is fp32 (converted to bf16 in regs).
__device__ __forceinline__ v16bf load_a16_f32(const float* __restrict__ A, int lda,
                                              int lane, int k0) {
  int r  = lane & 15;
  int kh = (lane >> 4) * 8;
  const float* p = A + (size_t)r * lda + k0 + kh;
  v16bf out;
#pragma unroll
  for (int i = 0; i < 8; ++i) { out[i] = f2bf(p[i]); out[i + 8] = f2bf(p[i + 16]); }
  return out;
}

// B fragment: 32x16 bf16, with B stored N-major (Bt[n][k], leading dim ldb).
// Lane L: col n = L&15, khalf = (L>>4)*8. Element e: e<8: k = khalf + e, e>=8: k = 16 + khalf + (e-8).
__device__ __forceinline__ v16bf load_bt16(const bf16_t* __restrict__ Bt, int ldb,
                                           int lane, int k0) {
  int n  = lane & 15;
  int kb = (lane >> 4) * 8;
  const bf16_t* p = Bt + (size_t)n * ldb + k0 + kb;
  bf16x8 lo = *(const bf16x8*)(p);
  bf16x8 hi = *(const bf16x8*)(p + 16);
  v16bf out;
#pragma unroll
  for (int i = 0; i < 8; ++i) { out[i] = lo[i]; out[i + 8] = hi[i]; }
  return out;
}

// ---------------------------------------------------------------- converts
__global__ __launch_bounds__(256) void cvt_f32_bf16(const float* __restrict__ src,
                                                    bf16_t* __restrict__ dst,
                                                    size_t n) {
  size_t i = (size_t)blockIdx.x * 256 + threadIdx.x;
  if (i < n) dst[i] = f2bf(src[i]);
}

// W: (K x N) row-major fp32  ->  Wt: (N x K) row-major bf16
__global__ __launch_bounds__(256) void transpose_cvt(const float* __restrict__ W,
                                                     bf16_t* __restrict__ Wt,
                                                     int K, int N) {
  size_t i = (size_t)blockIdx.x * 256 + threadIdx.x;
  if (i >= (size_t)K * N) return;
  int k = (int)(i / N);
  int n = (int)(i % N);
  Wt[(size_t)n * K + k] = f2bf(W[i]);
}

// ---------------------------------------------------------------- generic GEMM
// C(MxN fp32) = A(MxK bf16 row-major) * Bt(NxK bf16 row-major, i.e. B^T)
// 256 threads = 8 waves, wave tile 32x32, block tile 64(M) x 128(N).
// A/B tiles double-buffered in LDS, filled with async global->LDS DMA so the
// next K-tile's DMA overlaps the current tile's WMMA stream.
__global__ __launch_bounds__(256) void gemm_bf16_nt(const bf16_t* __restrict__ A,
                                                    const bf16_t* __restrict__ Bt,
                                                    float* __restrict__ C,
                                                    int M, int N, int K) {
  __shared__ bf16_t sA[2][64 * 32];    // [buf][row][k]  4 KB each
  __shared__ bf16_t sB[2][128 * 32];   // [buf][n][k]    8 KB each

  int t    = threadIdx.x;
  int lane = t & 31;
  int wave = t >> 5;
  int wm = wave >> 2;           // 0..1
  int wn = wave & 3;            // 0..3
  int mblk = blockIdx.y * 64;
  int nblk = blockIdx.x * 128;

  // per-thread 16B copy slots: A = 1 chunk, B = 2 chunks
  int cr = t >> 2;              // 0..63
  int cq = (t & 3) * 8;         // k offset 0/8/16/24
  const bf16_t* gA  = A  + (size_t)(mblk + cr) * K + cq;
  const bf16_t* gB0 = Bt + (size_t)(nblk + cr) * K + cq;
  const bf16_t* gB1 = Bt + (size_t)(nblk + 64 + cr) * K + cq;
  int lOff  = cr * 32 + cq;
  int lOff1 = (64 + cr) * 32 + cq;

  v8f acc00 = {}, acc01 = {}, acc10 = {}, acc11 = {};

  // prologue: stage K-tile 0 into buffer 0
  cp16_g2l(gA, &sA[0][lOff]);
  cp16_g2l(gB0, &sB[0][lOff]);
  cp16_g2l(gB1, &sB[0][lOff1]);

  int buf = 0;
  for (int k0 = 0; k0 < K; k0 += 32) {
    wait_async_lds();           // this wave's DMA into sX[buf] complete
    __syncthreads();            // every wave's tile visible

    if (k0 + 32 < K) {          // stage next tile into other buffer (overlaps WMMA)
      cp16_g2l(gA + k0 + 32, &sA[buf ^ 1][lOff]);
      cp16_g2l(gB0 + k0 + 32, &sB[buf ^ 1][lOff]);
      cp16_g2l(gB1 + k0 + 32, &sB[buf ^ 1][lOff1]);
    }

    v16bf a0 = load_a16(&sA[buf][(wm * 32) * 32], 32, lane, 0);
    v16bf a1 = load_a16(&sA[buf][(wm * 32 + 16) * 32], 32, lane, 0);
    v16bf b0 = load_bt16(&sB[buf][(wn * 32) * 32], 32, lane, 0);
    v16bf b1 = load_bt16(&sB[buf][(wn * 32 + 16) * 32], 32, lane, 0);
    acc00 = wmma_bf16(a0, b0, acc00);
    acc01 = wmma_bf16(a0, b1, acc01);
    acc10 = wmma_bf16(a1, b0, acc10);
    acc11 = wmma_bf16(a1, b1, acc11);

    buf ^= 1;
  }

  int m0 = mblk + wm * 32;
  int n0 = nblk + wn * 32;
  int cn = lane & 15;
  int cm = (lane >> 4) * 8;
#pragma unroll
  for (int r = 0; r < 8; ++r) {
    C[(size_t)(m0 + cm + r) * N + (n0 + cn)]           = acc00[r];
    C[(size_t)(m0 + cm + r) * N + (n0 + 16 + cn)]      = acc01[r];
    C[(size_t)(m0 + 16 + cm + r) * N + (n0 + cn)]      = acc10[r];
    C[(size_t)(m0 + 16 + cm + r) * N + (n0 + 16 + cn)] = acc11[r];
  }
}

// ---------------------------------------------------------------- RoPE reshape
// raw: [B*S, nheads*HD] fp32  ->  dst: [B, nheads, S, HD] bf16 with RoPE applied
__global__ __launch_bounds__(256) void rope_reshape(const float* __restrict__ raw,
                                                    const float* __restrict__ cosT,
                                                    const float* __restrict__ sinT,
                                                    bf16_t* __restrict__ dst,
                                                    int nheads, size_t total) {
  size_t i = (size_t)blockIdx.x * 256 + threadIdx.x;
  if (i >= total) return;
  int d = (int)(i % HD_);
  size_t t = i / HD_;
  int s = (int)(t % S_);
  size_t t2 = t / S_;
  int hh = (int)(t2 % nheads);
  int b  = (int)(t2 / nheads);

  const float* row = raw + ((size_t)b * S_ + s) * (size_t)(nheads * HD_) + (size_t)hh * HD_;
  float x = row[d];
  float other = (d < HD_ / 2) ? -row[d + HD_ / 2] : row[d - HD_ / 2];
  size_t cidx = ((size_t)b * S_ + s) * HD_ + d;
  dst[i] = f2bf(x * cosT[cidx] + other * sinT[cidx]);
}

// Vraw: [B*S, NKV*HD] fp32 -> Vt: [B, NKV, HD, S] bf16 (d-major for WMMA B loads)
__global__ __launch_bounds__(256) void v_reshape(const float* __restrict__ raw,
                                                 bf16_t* __restrict__ Vt,
                                                 size_t total) {
  size_t i = (size_t)blockIdx.x * 256 + threadIdx.x;
  if (i >= total) return;
  int s = (int)(i % S_);
  size_t t = i / S_;
  int d = (int)(t % HD_);
  size_t t2 = t / HD_;
  int kv = (int)(t2 % NKV_);
  int b  = (int)(t2 / NKV_);
  Vt[i] = f2bf(raw[((size_t)b * S_ + s) * (size_t)(NKV_ * HD_) + (size_t)kv * HD_ + d]);
}

// ---------------------------------------------------------------- scores
// W[b,h,q,k] = scale * Q(b,h)·K(b,kv)^T + mask[b,q,k]   (pre-softmax, fp32)
__global__ __launch_bounds__(256) void scores_kernel(const bf16_t* __restrict__ Q,
                                                     const bf16_t* __restrict__ Kc,
                                                     const float* __restrict__ mask,
                                                     float* __restrict__ W) {
  int bh = blockIdx.z;
  int b  = bh / NH_;
  int h  = bh % NH_;
  int kv = h / GROUPS_;
  const bf16_t* Qm = Q  + (size_t)bh * S_ * HD_;
  const bf16_t* Km = Kc + ((size_t)b * NKV_ + kv) * S_ * HD_;
  float* Wm        = W  + (size_t)bh * S_ * S_;
  const float* Mm  = mask + (size_t)b * S_ * S_;

  int lane = threadIdx.x & 31;
  int wave = threadIdx.x >> 5;
  int m0 = blockIdx.y * 64 + (wave >> 2) * 32;   // query rows
  int n0 = blockIdx.x * 128 + (wave & 3) * 32;   // key cols

  const bf16_t* A0 = Qm + (size_t)m0 * HD_;
  const bf16_t* A1 = Qm + (size_t)(m0 + 16) * HD_;
  const bf16_t* B0 = Km + (size_t)n0 * HD_;       // K stored (s,d): acts as B^T
  const bf16_t* B1 = Km + (size_t)(n0 + 16) * HD_;

  v8f acc00 = {}, acc01 = {}, acc10 = {}, acc11 = {};
#pragma unroll
  for (int k0 = 0; k0 < HD_; k0 += 32) {
    v16bf a0 = load_a16(A0, HD_, lane, k0);
    v16bf a1 = load_a16(A1, HD_, lane, k0);
    v16bf b0 = load_bt16(B0, HD_, lane, k0);
    v16bf b1 = load_bt16(B1, HD_, lane, k0);
    acc00 = wmma_bf16(a0, b0, acc00);
    acc01 = wmma_bf16(a0, b1, acc01);
    acc10 = wmma_bf16(a1, b0, acc10);
    acc11 = wmma_bf16(a1, b1, acc11);
  }

  int cn = lane & 15;
  int cm = (lane >> 4) * 8;
#pragma unroll
  for (int r = 0; r < 8; ++r) {
    size_t r0 = (size_t)(m0 + cm + r) * S_;
    size_t r1 = (size_t)(m0 + 16 + cm + r) * S_;
    Wm[r0 + n0 + cn]      = acc00[r] * SCALING_ + Mm[r0 + n0 + cn];
    Wm[r0 + n0 + 16 + cn] = acc01[r] * SCALING_ + Mm[r0 + n0 + 16 + cn];
    Wm[r1 + n0 + cn]      = acc10[r] * SCALING_ + Mm[r1 + n0 + cn];
    Wm[r1 + n0 + 16 + cn] = acc11[r] * SCALING_ + Mm[r1 + n0 + 16 + cn];
  }
}

// ---------------------------------------------------------------- softmax (rows of length S)
__global__ __launch_bounds__(256) void softmax_kernel(float* __restrict__ W) {
  __shared__ float red[256];
  float* p = W + (size_t)blockIdx.x * S_;
  int t = threadIdx.x;

  float m = -3.4e38f;
  for (int i = t; i < S_; i += 256) m = fmaxf(m, p[i]);
  red[t] = m; __syncthreads();
  for (int s = 128; s > 0; s >>= 1) { if (t < s) red[t] = fmaxf(red[t], red[t + s]); __syncthreads(); }
  float mx = red[0]; __syncthreads();

  float sum = 0.f;
  for (int i = t; i < S_; i += 256) { float e = __expf(p[i] - mx); p[i] = e; sum += e; }
  red[t] = sum; __syncthreads();
  for (int s = 128; s > 0; s >>= 1) { if (t < s) red[t] += red[t + s]; __syncthreads(); }
  float inv = 1.f / red[0];

  for (int i = t; i < S_; i += 256) p[i] *= inv;
}

// ---------------------------------------------------------------- AV
// Attn[b,q,h*HD+d] = sum_k P[b,h,q,k] * V[b,kv,k,d]; P fp32 (bf16 on load),
// Vt stored (b,kv,d,s) so B loads are contiguous. Causal tile skip: P==0 for k>q.
__global__ __launch_bounds__(256) void av_kernel(const float* __restrict__ P,
                                                 const bf16_t* __restrict__ Vt,
                                                 bf16_t* __restrict__ Attn) {
  int bh = blockIdx.z;
  int b  = bh / NH_;
  int h  = bh % NH_;
  int kv = h / GROUPS_;
  const float*  Pm = P  + (size_t)bh * S_ * S_;
  const bf16_t* Vm = Vt + ((size_t)b * NKV_ + kv) * HD_ * S_;   // [d][s]

  int lane = threadIdx.x & 31;
  int wave = threadIdx.x >> 5;
  int m0 = blockIdx.y * 64 + (wave >> 2) * 32;   // query rows
  int n0 = (wave & 3) * 32;                      // head-dim cols (HD=128 = whole block)

  const float*  A0 = Pm + (size_t)m0 * S_;
  const float*  A1 = Pm + (size_t)(m0 + 16) * S_;
  const bf16_t* B0 = Vm + (size_t)n0 * S_;
  const bf16_t* B1 = Vm + (size_t)(n0 + 16) * S_;

  int kend = blockIdx.y * 64 + 64;               // keys > max q in block contribute 0
  if (kend > S_) kend = S_;

  v8f acc00 = {}, acc01 = {}, acc10 = {}, acc11 = {};
  for (int k0 = 0; k0 < kend; k0 += 32) {
    v16bf a0 = load_a16_f32(A0, S_, lane, k0);
    v16bf a1 = load_a16_f32(A1, S_, lane, k0);
    v16bf b0 = load_bt16(B0, S_, lane, k0);
    v16bf b1 = load_bt16(B1, S_, lane, k0);
    acc00 = wmma_bf16(a0, b0, acc00);
    acc01 = wmma_bf16(a0, b1, acc01);
    acc10 = wmma_bf16(a1, b0, acc10);
    acc11 = wmma_bf16(a1, b1, acc11);
  }

  int cn = lane & 15;
  int cm = (lane >> 4) * 8;
#pragma unroll
  for (int r = 0; r < 8; ++r) {
    size_t row0 = ((size_t)b * S_ + (m0 + cm + r)) * (size_t)(NH_ * HD_) + (size_t)h * HD_;
    size_t row1 = ((size_t)b * S_ + (m0 + 16 + cm + r)) * (size_t)(NH_ * HD_) + (size_t)h * HD_;
    Attn[row0 + n0 + cn]      = f2bf(acc00[r]);
    Attn[row0 + n0 + 16 + cn] = f2bf(acc01[r]);
    Attn[row1 + n0 + cn]      = f2bf(acc10[r]);
    Attn[row1 + n0 + 16 + cn] = f2bf(acc11[r]);
  }
}

// ---------------------------------------------------------------- host
extern "C" void kernel_launch(void* const* d_in, const int* in_sizes, int n_in,
                              void* d_out, int out_size, void* d_ws, size_t ws_size,
                              hipStream_t stream) {
  const float* hidden = (const float*)d_in[0];
  const float* cosT   = (const float*)d_in[1];
  const float* sinT   = (const float*)d_in[2];
  const float* mask   = (const float*)d_in[3];
  const float* Wq     = (const float*)d_in[4];
  const float* Wk     = (const float*)d_in[5];
  const float* Wv     = (const float*)d_in[6];
  const float* Wo     = (const float*)d_in[7];

  float* out      = (float*)d_out;
  float* attn_out = out;                                   // (B,S,HID)
  float* weights  = out + (size_t)B_ * S_ * HID_;          // (B,NH,S,S)

  char* ws = (char*)d_ws;
  size_t off = 0;
  auto take = [&](size_t bytes) -> char* {
    size_t o = off;
    off += (bytes + 255) & ~(size_t)255;
    return ws + o;
  };

  const size_t MROWS = (size_t)B_ * S_;                    // 4096
  bf16_t* hidden_bf = (bf16_t*)take(MROWS * HID_ * 2);
  bf16_t* WqT = (bf16_t*)take((size_t)HID_ * (NH_ * HD_) * 2);
  bf16_t* WkT = (bf16_t*)take((size_t)HID_ * (NKV_ * HD_) * 2);
  bf16_t* WvT = (bf16_t*)take((size_t)HID_ * (NKV_ * HD_) * 2);
  bf16_t* WoT = (bf16_t*)take((size_t)(NH_ * HD_) * HID_ * 2);
  float*  Qraw = (float*)take(MROWS * (NH_ * HD_) * 4);
  float*  Kraw = (float*)take(MROWS * (NKV_ * HD_) * 4);
  float*  Vraw = (float*)take(MROWS * (NKV_ * HD_) * 4);
  bf16_t* Qbf  = (bf16_t*)take((size_t)B_ * NH_ * S_ * HD_ * 2);
  bf16_t* Kbf  = (bf16_t*)take((size_t)B_ * NKV_ * S_ * HD_ * 2);
  bf16_t* Vtbf = (bf16_t*)take((size_t)B_ * NKV_ * HD_ * S_ * 2);
  bf16_t* AttnBf = (bf16_t*)take(MROWS * (size_t)(NH_ * HD_) * 2);

  // 1) precision convert + weight transposes
  {
    size_t n = MROWS * HID_;
    cvt_f32_bf16<<<(unsigned)((n + 255) / 256), 256, 0, stream>>>(hidden, hidden_bf, n);
  }
  transpose_cvt<<<(HID_ * NH_ * HD_) / 256, 256, 0, stream>>>(Wq, WqT, HID_, NH_ * HD_);
  transpose_cvt<<<(HID_ * NKV_ * HD_) / 256, 256, 0, stream>>>(Wk, WkT, HID_, NKV_ * HD_);
  transpose_cvt<<<(HID_ * NKV_ * HD_) / 256, 256, 0, stream>>>(Wv, WvT, HID_, NKV_ * HD_);
  transpose_cvt<<<((NH_ * HD_) * HID_) / 256, 256, 0, stream>>>(Wo, WoT, NH_ * HD_, HID_);

  // 2) QKV projection GEMMs (async-LDS staged WMMA bf16 -> fp32)
  gemm_bf16_nt<<<dim3((NH_ * HD_) / 128, (unsigned)(MROWS / 64)), 256, 0, stream>>>(
      hidden_bf, WqT, Qraw, (int)MROWS, NH_ * HD_, HID_);
  gemm_bf16_nt<<<dim3((NKV_ * HD_) / 128, (unsigned)(MROWS / 64)), 256, 0, stream>>>(
      hidden_bf, WkT, Kraw, (int)MROWS, NKV_ * HD_, HID_);
  gemm_bf16_nt<<<dim3((NKV_ * HD_) / 128, (unsigned)(MROWS / 64)), 256, 0, stream>>>(
      hidden_bf, WvT, Vraw, (int)MROWS, NKV_ * HD_, HID_);

  // 3) RoPE + layout reshapes
  {
    size_t nq = (size_t)B_ * NH_ * S_ * HD_;
    rope_reshape<<<(unsigned)((nq + 255) / 256), 256, 0, stream>>>(Qraw, cosT, sinT, Qbf, NH_, nq);
    size_t nk = (size_t)B_ * NKV_ * S_ * HD_;
    rope_reshape<<<(unsigned)((nk + 255) / 256), 256, 0, stream>>>(Kraw, cosT, sinT, Kbf, NKV_, nk);
    v_reshape<<<(unsigned)((nk + 255) / 256), 256, 0, stream>>>(Vraw, Vtbf, nk);
  }

  // 4) scores = scale*QK^T + mask  -> weights region (pre-softmax)
  scores_kernel<<<dim3(S_ / 128, S_ / 64, B_ * NH_), 256, 0, stream>>>(Qbf, Kbf, mask, weights);

  // 5) softmax over each row of length S (in place, becomes attn_weights output)
  softmax_kernel<<<B_ * NH_ * S_, 256, 0, stream>>>(weights);

  // 6) context = P @ V (causal tile skip)
  av_kernel<<<dim3(1, S_ / 64, B_ * NH_), 256, 0, stream>>>(weights, Vtbf, AttnBf);

  // 7) output projection -> attn_output region
  gemm_bf16_nt<<<dim3(HID_ / 128, (unsigned)(MROWS / 64)), 256, 0, stream>>>(
      AttnBf, WoT, attn_out, (int)MROWS, HID_, NH_ * HD_);
}
